// CoAttentionFusion_78219944395016
// MI455X (gfx1250) — matrix operand order repaired and checked by reading, not compiled, not run on hardware
//
#include <hip/hip_runtime.h>
#include <math.h>

#define D_MODEL  1024
#define N_HEADS  16
#define HEAD_DIM 64
#define BATCH    4
#define SEQ      2048
#define BT       (BATCH * SEQ)   // 8192 rows

typedef __bf16 bf16_t;
typedef __attribute__((ext_vector_type(16))) __bf16 v16bf;
typedef __attribute__((ext_vector_type(8)))  float  v8f;
typedef unsigned int u32x4 __attribute__((ext_vector_type(4)));
typedef int   i32x4 __attribute__((ext_vector_type(4)));
typedef int   i32x8 __attribute__((ext_vector_type(8)));
typedef float f32x4 __attribute__((ext_vector_type(4)));

#if defined(__gfx1250__) && __has_builtin(__builtin_amdgcn_tensor_load_to_lds)
#define USE_TDM 1
#else
#define USE_TDM 0
#endif

union Frag {            // 32-byte fragment shuttle (trivial types only)
    u32x4  q[2];
    v16bf  v;
    bf16_t e[16];
};

__device__ __forceinline__ v8f wmma_bf16(v16bf a, v16bf b, v8f c) {
    // D = A(16x32 bf16) * B(32x16 bf16) + C(16x16 f32)
    return __builtin_amdgcn_wmma_f32_16x16x32_bf16(
        /*neg_a=*/false, a, /*neg_b=*/false, b,
        /*c_mod=*/(short)0, c, /*reuse_a=*/false, /*reuse_b=*/false);
}

// ---------------------------------------------------------------------------
// fp32 -> bf16 cast (grid-stride)
// ---------------------------------------------------------------------------
__global__ void cast_f32_bf16(const float* __restrict__ in,
                              bf16_t* __restrict__ out, long n) {
    long i = (long)blockIdx.x * blockDim.x + threadIdx.x;
    long stride = (long)gridDim.x * blockDim.x;
    for (; i < n; i += stride) out[i] = (bf16_t)in[i];
}

// ---------------------------------------------------------------------------
// C = A[M,K](bf16) @ W[K,N](bf16, row-major "in,out") + bias
// MODE 0: fp32 out   MODE 1: bf16 out   MODE 2: bf16 out with exact GELU
// Block: 256 threads = 8 waves (2x4), tile 128x128, k-step 32.
// ---------------------------------------------------------------------------
template<int MODE>
__global__ __launch_bounds__(256)
void gemm_wmma(const bf16_t* __restrict__ A, const bf16_t* __restrict__ W,
               const float* __restrict__ bias, void* __restrict__ outp,
               int M, int N, int K) {
    __shared__ __align__(16) bf16_t lA[128 * 32];   // [m][k]
    __shared__ __align__(16) bf16_t lB[128 * 32];   // [n][k] (transposed W)

    const int tid  = threadIdx.x;
    const int wave = tid >> 5;          // 0..7
    const int lane = tid & 31;
    const int hilo = lane >> 4;         // K-half select per WMMA layout
    const int l16  = lane & 15;
    const int wm   = wave >> 2;         // 0..1  (M direction)
    const int wn   = wave & 3;          // 0..3  (N direction)
    const int m0   = blockIdx.y * 128;
    const int n0   = blockIdx.x * 128;

    v8f zero = {};
    v8f acc[4][2];
#pragma unroll
    for (int i = 0; i < 4; ++i)
#pragma unroll
        for (int j = 0; j < 2; ++j) acc[i][j] = zero;

    const int ar = tid >> 1;            // A stage: row 0..127
    const int ak = (tid & 1) * 16;      //          k 0 or 16
    const int bk = tid >> 3;            // B stage: k 0..31
    const int bn = (tid & 7) * 16;      //          n base 0..112

    for (int k0 = 0; k0 < K; k0 += 32) {
        {   // stage A tile (row-major, 32B per thread)
            const bf16_t* src = A + (size_t)(m0 + ar) * K + k0 + ak;
            *(u32x4*)&lA[ar * 32 + ak]     = *(const u32x4*)src;
            *(u32x4*)&lA[ar * 32 + ak + 8] = *(const u32x4*)(src + 8);
            if (k0 + 32 < K) __builtin_prefetch(src + 32, 0, 1);
        }
        {   // stage W tile transposed -> lB[n][k]
            const bf16_t* src = W + (size_t)(k0 + bk) * N + n0 + bn;
            Frag tw;
            tw.q[0] = *(const u32x4*)src;
            tw.q[1] = *(const u32x4*)(src + 8);
#pragma unroll
            for (int i = 0; i < 16; ++i) lB[(bn + i) * 32 + bk] = tw.e[i];
        }
        __syncthreads();

        v16bf afr[4];
#pragma unroll
        for (int i = 0; i < 4; ++i) {   // A frags: lane=M row, half-wave = K+8
            Frag f;
            const bf16_t* p = &lA[(wm * 64 + i * 16 + l16) * 32 + hilo * 8];
            f.q[0] = *(const u32x4*)p;
            f.q[1] = *(const u32x4*)(p + 16);
            afr[i] = f.v;
        }
        v16bf bfr[2];
#pragma unroll
        for (int j = 0; j < 2; ++j) {   // B frags: lane=N col, half-wave = K+8
            Frag f;
            const bf16_t* p = &lB[(wn * 32 + j * 16 + l16) * 32 + hilo * 8];
            f.q[0] = *(const u32x4*)p;
            f.q[1] = *(const u32x4*)(p + 16);
            bfr[j] = f.v;
        }
#pragma unroll
        for (int i = 0; i < 4; ++i)
#pragma unroll
            for (int j = 0; j < 2; ++j)
                acc[i][j] = wmma_bf16(afr[i], bfr[j], acc[i][j]);
        __syncthreads();
    }

    // epilogue: C layout lane=N, vgpr r = M row (half-wave = M+8)
#pragma unroll
    for (int i = 0; i < 4; ++i) {
        const int rowb = m0 + wm * 64 + i * 16 + 8 * hilo;
#pragma unroll
        for (int j = 0; j < 2; ++j) {
            const int col = n0 + wn * 32 + j * 16 + l16;
            const float bv = bias ? bias[col] : 0.0f;
#pragma unroll
            for (int r = 0; r < 8; ++r) {
                float v = acc[i][j][r] + bv;
                if (MODE == 2) v = 0.5f * v * (1.0f + erff(v * 0.70710678118f));
                const size_t idx = (size_t)(rowb + r) * N + col;
                if (MODE == 0) ((float*)outp)[idx] = v;
                else           ((bf16_t*)outp)[idx] = (bf16_t)v;
            }
        }
    }
}

// ---------------------------------------------------------------------------
// Flash attention: one block (4 waves) per 64 query rows per (b,h).
// Q,K,V,Ctx are [B,T,D] bf16 with head h at columns h*64..h*64+63.
// K tile staged via the Tensor Data Mover (D# descriptor, TENSORcnt).
// ---------------------------------------------------------------------------
__global__ __launch_bounds__(128)
void attn_wmma(const bf16_t* __restrict__ Q, const bf16_t* __restrict__ Kg,
               const bf16_t* __restrict__ V, bf16_t* __restrict__ Ctx) {
    __shared__ __align__(16) bf16_t kT[32 * 64];       // [key][hd]
    __shared__ __align__(16) bf16_t vT[64 * 32];       // [hd][key] (transposed)
    __shared__ __align__(16) bf16_t pT[4 * 16 * 32];   // per-wave P staging

    const int tid  = threadIdx.x;
    const int wave = tid >> 5;          // 0..3 -> q sub-tile
    const int lane = tid & 31;
    const int hilo = lane >> 4;
    const int l16  = lane & 15;

    const int bh = blockIdx.y;
    const int b  = bh / N_HEADS;
    const int h  = bh % N_HEADS;
    const int q0 = blockIdx.x * 64;

    // Q fragments for this wave's 16 rows (resident all loop)
    v16bf qa[2];
    {
        const bf16_t* qrow =
            Q + ((size_t)b * SEQ + q0 + wave * 16 + l16) * D_MODEL + h * HEAD_DIM;
#pragma unroll
        for (int ks = 0; ks < 2; ++ks) {
            Frag f;
            f.q[0] = *(const u32x4*)(qrow + ks * 32 + hilo * 8);
            f.q[1] = *(const u32x4*)(qrow + ks * 32 + 16 + hilo * 8);
            qa[ks] = f.v;
        }
    }

    float mrun[8], lrun[8];
#pragma unroll
    for (int r = 0; r < 8; ++r) { mrun[r] = -3.0e38f; lrun[r] = 0.0f; }
    v8f zero = {};
    v8f o[4];
#pragma unroll
    for (int t = 0; t < 4; ++t) o[t] = zero;

    const int sr = tid >> 2;            // staging row 0..31
    const int sc = (tid & 3) * 16;      // staging col chunk

    for (int kb = 0; kb < SEQ; kb += 32) {
#if USE_TDM
        if (wave == 0) {
            // Tensor DMA: 32x64 bf16 tile, row stride D_MODEL, into kT.
            const unsigned long long ga = (unsigned long long)(const void*)
                (Kg + ((size_t)b * SEQ + kb) * D_MODEL + h * HEAD_DIM);
            u32x4 g0; i32x8 g1; i32x4 gz4 = {}; i32x8 gz8 = {};
            g0[0] = 1u;                                   // count=1, user mode
            g0[1] = (unsigned)(size_t)(void*)kT;          // lds_addr
            g0[2] = (unsigned)(ga & 0xFFFFFFFFull);       // global_addr lo
            g0[3] = (unsigned)((ga >> 32) & 0x01FFFFFFull) | (2u << 30); // hi|type=2
            g1[0] = (int)(1u << 16);                      // data_size = 2B
            g1[1] = (int)((HEAD_DIM & 0xFFFF) << 16);     // tensor_dim0 lo
            g1[2] = (int)((32u & 0xFFFFu) << 16);         // td0 hi=0 | tensor_dim1=32
            g1[3] = (int)((HEAD_DIM & 0xFFFF) << 16);     // td1 hi=0 | tile_dim0=64
            g1[4] = (int)32;                              // tile_dim1=32, tile_dim2=0
            g1[5] = (int)D_MODEL;                         // tensor_dim0_stride lo
            g1[6] = 0;                                    // stride0 hi | stride1 lo
            g1[7] = 0;                                    // stride1 hi
            __builtin_amdgcn_tensor_load_to_lds(g0, g1, gz4, gz4, gz8, 0);
        }
        __builtin_amdgcn_s_wait_tensorcnt(0);   // no-op for non-issuing waves
#else
        {   // stage K block row-major
            const bf16_t* src =
                Kg + ((size_t)b * SEQ + kb + sr) * D_MODEL + h * HEAD_DIM + sc;
            *(u32x4*)&kT[sr * 64 + sc]     = *(const u32x4*)src;
            *(u32x4*)&kT[sr * 64 + sc + 8] = *(const u32x4*)(src + 8);
        }
#endif
        {   // stage V block transposed (TDM cannot transpose)
            const bf16_t* src =
                V + ((size_t)b * SEQ + kb + sr) * D_MODEL + h * HEAD_DIM + sc;
            Frag tv;
            tv.q[0] = *(const u32x4*)src;
            tv.q[1] = *(const u32x4*)(src + 8);
#pragma unroll
            for (int i = 0; i < 16; ++i) vT[(sc + i) * 32 + sr] = tv.e[i];
        }
        __syncthreads();

        // S = Q * K^T for two 16-key sub-blocks (contract hd=64 in 2 WMMAs)
        v8f st[2];
#pragma unroll
        for (int j = 0; j < 2; ++j) {
            v8f s = zero;
#pragma unroll
            for (int ks = 0; ks < 2; ++ks) {
                Frag f;
                const bf16_t* p = &kT[(j * 16 + l16) * 64 + ks * 32 + hilo * 8];
                f.q[0] = *(const u32x4*)p;
                f.q[1] = *(const u32x4*)(p + 16);
                s = wmma_bf16(qa[ks], f.v, s);
            }
            st[j] = s * 0.125f;   // 1/sqrt(64)
        }

        // online softmax; row stats replicated across each half-wave (width 16)
        float fac[8];
#pragma unroll
        for (int r = 0; r < 8; ++r) {
            float bm = fmaxf(st[0][r], st[1][r]);
            bm = fmaxf(bm, __shfl_xor(bm, 1, 16));
            bm = fmaxf(bm, __shfl_xor(bm, 2, 16));
            bm = fmaxf(bm, __shfl_xor(bm, 4, 16));
            bm = fmaxf(bm, __shfl_xor(bm, 8, 16));
            float mn = fmaxf(mrun[r], bm);
            fac[r] = __expf(mrun[r] - mn);
            mrun[r] = mn;
            lrun[r] *= fac[r];
        }
#pragma unroll
        for (int t = 0; t < 4; ++t)
#pragma unroll
            for (int r = 0; r < 8; ++r) o[t][r] *= fac[r];

#pragma unroll
        for (int r = 0; r < 8; ++r) {
            float p0 = __expf(st[0][r] - mrun[r]);
            float p1 = __expf(st[1][r] - mrun[r]);
            st[0][r] = p0; st[1][r] = p1;
            float rs = p0 + p1;
            rs += __shfl_xor(rs, 1, 16);
            rs += __shfl_xor(rs, 2, 16);
            rs += __shfl_xor(rs, 4, 16);
            rs += __shfl_xor(rs, 8, 16);
            lrun[r] += rs;
        }

        // C-layout fp32 -> A-layout bf16 via per-wave LDS pad (in-order DS)
        bf16_t* pw = &pT[wave * 16 * 32];
#pragma unroll
        for (int r = 0; r < 8; ++r) {
            pw[(r + 8 * hilo) * 32 + l16]      = (bf16_t)st[0][r];
            pw[(r + 8 * hilo) * 32 + 16 + l16] = (bf16_t)st[1][r];
        }
        Frag fp;
        fp.q[0] = *(const u32x4*)&pw[l16 * 32 + hilo * 8];
        fp.q[1] = *(const u32x4*)&pw[l16 * 32 + 16 + hilo * 8];

        // O += P(16x32) * V(32x64), 4 hd sub-tiles
#pragma unroll
        for (int t = 0; t < 4; ++t) {
            Frag fv;
            const bf16_t* p = &vT[(t * 16 + l16) * 32 + hilo * 8];
            fv.q[0] = *(const u32x4*)p;
            fv.q[1] = *(const u32x4*)(p + 16);
            o[t] = wmma_bf16(fp.v, fv.v, o[t]);
        }
        __syncthreads();
    }

    // normalize and store context (bf16, next GEMM input)
#pragma unroll
    for (int t = 0; t < 4; ++t)
#pragma unroll
        for (int r = 0; r < 8; ++r) {
            const int row = q0 + wave * 16 + r + 8 * hilo;
            const int col = h * HEAD_DIM + t * 16 + l16;
            Ctx[((size_t)b * SEQ + row) * D_MODEL + col] =
                (bf16_t)(o[t][r] / lrun[r]);
        }
}

// ---------------------------------------------------------------------------
// LayerNorm over D=1024; optional fp32 residual; fp32 or strided-bf16 output.
// One 256-thread block per row, 4 elements per thread.
// ---------------------------------------------------------------------------
template<bool BF16OUT>
__global__ __launch_bounds__(256)
void layernorm_k(const float* __restrict__ x, const float* __restrict__ resid,
                 const float* __restrict__ g, const float* __restrict__ beta,
                 void* __restrict__ outp, int ldOut) {
    const int row = blockIdx.x;
    const int tid = threadIdx.x;
    const int c0  = tid * 4;

    f32x4 xv = *(const f32x4*)(x + (size_t)row * D_MODEL + c0);
    if (resid) {
        f32x4 rv = *(const f32x4*)(resid + (size_t)row * D_MODEL + c0);
        xv = xv + rv;
    }
    float s = xv[0] + xv[1] + xv[2] + xv[3];
    float q = xv[0]*xv[0] + xv[1]*xv[1] + xv[2]*xv[2] + xv[3]*xv[3];
    for (int m = 16; m >= 1; m >>= 1) {
        s += __shfl_xor(s, m, 32);
        q += __shfl_xor(q, m, 32);
    }
    __shared__ float rs[8], rq[8];
    const int wid = tid >> 5, lid = tid & 31;
    if (lid == 0) { rs[wid] = s; rq[wid] = q; }
    __syncthreads();
    if (tid < 32) {
        s = (lid < 8) ? rs[lid] : 0.0f;
        q = (lid < 8) ? rq[lid] : 0.0f;
        for (int m = 4; m >= 1; m >>= 1) {
            s += __shfl_xor(s, m, 32);
            q += __shfl_xor(q, m, 32);
        }
        if (lid == 0) { rs[0] = s; rq[0] = q; }
    }
    __syncthreads();
    const float mean = rs[0] * (1.0f / D_MODEL);
    const float var  = rq[0] * (1.0f / D_MODEL) - mean * mean;
    const float inv  = rsqrtf(var + 1e-5f);
#pragma unroll
    for (int i = 0; i < 4; ++i) {
        const int c = c0 + i;
        const float y = (xv[i] - mean) * inv * g[c] + beta[c];
        if (BF16OUT) ((bf16_t*)outp)[(size_t)row * ldOut + c] = (bf16_t)y;
        else         ((float*)outp)[(size_t)row * ldOut + c]  = y;
    }
}

// ---------------------------------------------------------------------------
extern "C" void kernel_launch(void* const* d_in, const int* in_sizes, int n_in,
                              void* d_out, int out_size, void* d_ws, size_t ws_size,
                              hipStream_t stream) {
    (void)in_sizes; (void)n_in; (void)out_size; (void)ws_size;
    const float* temporal = (const float*)d_in[0];
    const float* feature  = (const float*)d_in[1];
    const float* Wf32[8]; const float* Bvec[8];
    for (int i = 0; i < 8; ++i) {            // qt,kf,vf,qf,kt,vt,ot,of
        Wf32[i] = (const float*)d_in[2 + 2 * i];
        Bvec[i] = (const float*)d_in[3 + 2 * i];
    }
    const float* fus1_w = (const float*)d_in[18];
    const float* fus1_b = (const float*)d_in[19];
    const float* fus2_w = (const float*)d_in[20];
    const float* fus2_b = (const float*)d_in[21];
    const float* ln_fus_w = (const float*)d_in[22];
    const float* ln_fus_b = (const float*)d_in[23];
    const float* ln_t_w = (const float*)d_in[24];
    const float* ln_t_b = (const float*)d_in[25];
    const float* ln_f_w = (const float*)d_in[26];
    const float* ln_f_b = (const float*)d_in[27];

    char* ws = (char*)d_ws;
    size_t off = 0;
    auto alloc = [&](size_t bytes) -> char* {
        char* p = ws + off;
        off = (off + bytes + 255) & ~(size_t)255;
        return p;
    };
    const size_t BTD = (size_t)BT * D_MODEL;
    const size_t DD  = (size_t)D_MODEL * D_MODEL;
    bf16_t* bXt  = (bf16_t*)alloc(BTD * 2);
    bf16_t* bXf  = (bf16_t*)alloc(BTD * 2);
    bf16_t* bW   = (bf16_t*)alloc(8 * DD * 2);
    bf16_t* bW1  = (bf16_t*)alloc(2 * DD * 2);
    bf16_t* bW2  = (bf16_t*)alloc(DD * 2);
    bf16_t* bQ   = (bf16_t*)alloc(BTD * 2);
    bf16_t* bK   = (bf16_t*)alloc(BTD * 2);
    bf16_t* bV   = (bf16_t*)alloc(BTD * 2);
    bf16_t* bCtx = (bf16_t*)alloc(BTD * 2);
    bf16_t* bComb= (bf16_t*)alloc(BTD * 2 * 2);   // [BT][2048]
    bf16_t* bH   = (bf16_t*)alloc(BTD * 2);
    float*  fTmp = (float*)alloc(BTD * 4);

    // ---- casts to bf16
    cast_f32_bf16<<<2048, 256, 0, stream>>>(temporal, bXt, (long)BTD);
    cast_f32_bf16<<<2048, 256, 0, stream>>>(feature,  bXf, (long)BTD);
    for (int i = 0; i < 8; ++i)
        cast_f32_bf16<<<512, 256, 0, stream>>>(Wf32[i], bW + i * DD, (long)DD);
    cast_f32_bf16<<<1024, 256, 0, stream>>>(fus1_w, bW1, (long)(2 * DD));
    cast_f32_bf16<<<512,  256, 0, stream>>>(fus2_w, bW2, (long)DD);

    dim3 ggrd(D_MODEL / 128, BT / 128);          // (8, 64)
    dim3 agrd(SEQ / 64, BATCH * N_HEADS);        // (32, 64)

    // ---- temporal -> feature cross attention
    gemm_wmma<1><<<ggrd, 256, 0, stream>>>(bXt, bW + 0 * DD, Bvec[0], bQ, BT, D_MODEL, D_MODEL);
    gemm_wmma<1><<<ggrd, 256, 0, stream>>>(bXf, bW + 1 * DD, Bvec[1], bK, BT, D_MODEL, D_MODEL);
    gemm_wmma<1><<<ggrd, 256, 0, stream>>>(bXf, bW + 2 * DD, Bvec[2], bV, BT, D_MODEL, D_MODEL);
    attn_wmma<<<agrd, 128, 0, stream>>>(bQ, bK, bV, bCtx);
    gemm_wmma<0><<<ggrd, 256, 0, stream>>>(bCtx, bW + 6 * DD, Bvec[6], fTmp, BT, D_MODEL, D_MODEL);
    layernorm_k<true><<<BT, 256, 0, stream>>>(fTmp, temporal, ln_t_w, ln_t_b,
                                              bComb, 2 * D_MODEL);

    // ---- feature -> temporal cross attention (scratch reuse)
    gemm_wmma<1><<<ggrd, 256, 0, stream>>>(bXf, bW + 3 * DD, Bvec[3], bQ, BT, D_MODEL, D_MODEL);
    gemm_wmma<1><<<ggrd, 256, 0, stream>>>(bXt, bW + 4 * DD, Bvec[4], bK, BT, D_MODEL, D_MODEL);
    gemm_wmma<1><<<ggrd, 256, 0, stream>>>(bXt, bW + 5 * DD, Bvec[5], bV, BT, D_MODEL, D_MODEL);
    attn_wmma<<<agrd, 128, 0, stream>>>(bQ, bK, bV, bCtx);
    gemm_wmma<0><<<ggrd, 256, 0, stream>>>(bCtx, bW + 7 * DD, Bvec[7], fTmp, BT, D_MODEL, D_MODEL);
    layernorm_k<true><<<BT, 256, 0, stream>>>(fTmp, feature, ln_f_w, ln_f_b,
                                              bComb + D_MODEL, 2 * D_MODEL);

    // ---- fusion MLP: gelu(comb @ fus1 + b) @ fus2 + b -> LN -> d_out (fp32)
    gemm_wmma<2><<<ggrd, 256, 0, stream>>>(bComb, bW1, fus1_b, bH, BT, D_MODEL, 2 * D_MODEL);
    gemm_wmma<0><<<ggrd, 256, 0, stream>>>(bH, bW2, fus2_b, fTmp, BT, D_MODEL, D_MODEL);
    layernorm_k<false><<<BT, 256, 0, stream>>>(fTmp, nullptr, ln_fus_w, ln_fus_b,
                                               d_out, D_MODEL);
}
